// GINGRURegressor_53626961658409
// MI455X (gfx1250) — matrix operand order repaired and checked by reading, not compiled, not run on hardware
//
#include <hip/hip_runtime.h>
#include <hip/hip_bf16.h>

// ---------------------------------------------------------------------------
// GIN(2 layers, fused per-window) + mean-pool + masked GRU + linear head
// CDNA5 / gfx1250: wave32, WMMA 16x16x32 bf16, CSR gather aggregation in LDS,
// bank-conflict-padded LDS tiles, weights staged to LDS as bf16 transposed.
// ---------------------------------------------------------------------------

#define NPW   128      // nodes per window
#define EPW   2048     // edges per window
#define NWIN  1024     // windows
#define DIM   128      // feature dim (IN_DIM == GNN_H == GRU_H)
#define BSUBJ 32       // subjects
#define E_TOT (NWIN * EPW)

#define FSTR  132      // feat row stride in f32   (pad: banks rotate by 4)
#define FSTR4 33       // feat row stride in float4
#define HSTR  136      // bf16 row stride (abf/wbf) (pad: banks rotate by 4)

typedef __attribute__((ext_vector_type(16))) __bf16 v16bf;
typedef __attribute__((ext_vector_type(8)))  float  v8f;

struct __align__(16) bf16x8 { __bf16 h[8]; };
struct __align__(8)  bf16x4 { __bf16 h[4]; };

// Build a 16-element bf16 fragment from two 16B LDS chunks at p and p+gap.
//   A (16x32, per ISA §7.12.2): gap=16  (K kb..+7 & kb+16..+23 per lane half)
//   B (32x16, transposed wT): gap=16 (same K split as A)
__device__ __forceinline__ v16bf load_frag(const __bf16* p, int gap) {
    union { v16bf v; bf16x8 c[2]; } u;
    u.c[0] = *reinterpret_cast<const bf16x8*>(p);
    u.c[1] = *reinterpret_cast<const bf16x8*>(p + gap);
    return u.v;
}

// ---------------------------------------------------------------------------
// 128x128x128 GEMM: out = relu(A @ W + bias).
// Stages W (global f32 [K][N]) -> wbf (LDS bf16 [N][K], stride HSTR), then
// 8 waves each compute a 2(M) x 4(N) block of 16x16 tiles; each B fragment
// feeds two WMMAs. All A/B reads precede the barrier, so outBF may alias Abuf.
// ---------------------------------------------------------------------------
__device__ __forceinline__ void gemm_relu_128(const __bf16* __restrict__ Abuf,
                                              const float*  __restrict__ Wglob,
                                              __bf16*       __restrict__ wbf,
                                              const float*  __restrict__ bias,
                                              __bf16*       __restrict__ outBF,
                                              float*        __restrict__ outF32,
                                              int tid) {
    // stage weights: coalesced f32 reads, bf16 transposed writes (2-way conflict)
    for (int i = tid; i < DIM * DIM; i += 256) {
        const int k = i >> 7, n = i & 127;
        wbf[n * HSTR + k] = (__bf16)Wglob[i];
    }
    __syncthreads();

    const int lane  = tid & 31;
    const int wave  = tid >> 5;
    const int mt0   = (wave & 3) * 2;        // first of 2 M-tiles
    const int ng    = (wave >> 2) * 4;       // first of 4 N-tiles
    const int rlo   = lane & 15;
    const int khalf = (lane < 16) ? 0 : 8;   // A-fragment K split
    const int koff  = (lane < 16) ? 0 : 16;  // unused (B fragment also uses khalf)
    const int rowHi = (lane < 16) ? 0 : 8;   // C/D row offset

    v16bf afrag[2][4];
#pragma unroll
    for (int mt = 0; mt < 2; ++mt) {
        const int row = (mt0 + mt) * 16 + rlo;
#pragma unroll
        for (int kk = 0; kk < 4; ++kk)
            afrag[mt][kk] = load_frag(Abuf + row * HSTR + kk * 32 + khalf, 16);
    }

    v8f acc[2][4];
#pragma unroll
    for (int jn = 0; jn < 4; ++jn) {
        const int col = (ng + jn) * 16 + rlo;
        v8f c0 = {0.f, 0.f, 0.f, 0.f, 0.f, 0.f, 0.f, 0.f};
        v8f c1 = {0.f, 0.f, 0.f, 0.f, 0.f, 0.f, 0.f, 0.f};
#pragma unroll
        for (int kk = 0; kk < 4; ++kk) {
            v16bf b = load_frag(wbf + col * HSTR + kk * 32 + khalf, 16);
            c0 = __builtin_amdgcn_wmma_f32_16x16x32_bf16(
                     false, afrag[0][kk], false, b, (short)0, c0, false, false);
            c1 = __builtin_amdgcn_wmma_f32_16x16x32_bf16(
                     false, afrag[1][kk], false, b, (short)0, c1, false, false);
        }
        const float bv = bias[col];
#pragma unroll
        for (int r = 0; r < 8; ++r) {
            c0[r] = fmaxf(c0[r] + bv, 0.f);
            c1[r] = fmaxf(c1[r] + bv, 0.f);
        }
        acc[0][jn] = c0;
        acc[1][jn] = c1;
    }

    __syncthreads();   // all Abuf/wbf reads complete before aliasing writes
#pragma unroll
    for (int mt = 0; mt < 2; ++mt) {
        const int mbase = (mt0 + mt) * 16 + rowHi;
#pragma unroll
        for (int jn = 0; jn < 4; ++jn) {
            const int col = (ng + jn) * 16 + rlo;
#pragma unroll
            for (int r = 0; r < 8; ++r) {
                const int m = mbase + r;
                const float v = acc[mt][jn][r];
                outBF[m * HSTR + col] = (__bf16)v;
                if (outF32) outF32[m * FSTR + col] = v;
            }
        }
    }
    __syncthreads();
}

// ---------------------------------------------------------------------------
// GIN aggregation, gather form (no atomics): for each dst row,
//   z[dst] = feat[dst] + sum_{e: dst} feat[src(e)]   -> abf (bf16)
// wave -> dst row (uniform rowptr/csr per wave), lane -> float4 dim group.
// ---------------------------------------------------------------------------
__device__ __forceinline__ void aggregate_csr(const float* __restrict__ feat,
                                              __bf16*      __restrict__ abf,
                                              const int*   __restrict__ rowptr,
                                              const int*   __restrict__ csr_src,
                                              int tid) {
    const int lane = tid & 31;
    const int wave = tid >> 5;
    const float4* frow = reinterpret_cast<const float4*>(feat);
    for (int dst = wave; dst < NPW; dst += 8) {
        float4 a = frow[dst * FSTR4 + lane];
        const int beg = rowptr[dst], end = rowptr[dst + 1];
        for (int e = beg; e < end; ++e) {
            const float4 v = frow[csr_src[e] * FSTR4 + lane];
            a.x += v.x; a.y += v.y; a.z += v.z; a.w += v.w;
        }
        bf16x4 o;
        o.h[0] = (__bf16)a.x; o.h[1] = (__bf16)a.y;
        o.h[2] = (__bf16)a.z; o.h[3] = (__bf16)a.w;
        *reinterpret_cast<bf16x4*>(&abf[dst * HSTR + lane * 4]) = o;
    }
    __syncthreads();
}

// ---------------------------------------------------------------------------
// Kernel 1: one workgroup (8 wave32) per window. CSR build -> GIN1 -> GIN2
// -> mean pool. Everything stays in LDS; x and edges are read once (edges 2x).
// ---------------------------------------------------------------------------
__global__ __launch_bounds__(256) void gin_window_kernel(
    const float* __restrict__ x,
    const int*   __restrict__ esrc,   // edge_index[0]
    const int*   __restrict__ edst,   // edge_index[1]
    const float* __restrict__ w1a, const float* __restrict__ b1a,
    const float* __restrict__ w1b, const float* __restrict__ b1b,
    const float* __restrict__ w2a, const float* __restrict__ b2a,
    const float* __restrict__ w2b, const float* __restrict__ b2b,
    float* __restrict__ g)            // [NWIN, DIM]
{
    __shared__ float  feat[NPW * FSTR];    // 66 KB: f32 features (padded)
    __shared__ __bf16 abf [NPW * HSTR];    // 34 KB: bf16 A-matrix (padded)
    __shared__ __bf16 wbf [NPW * HSTR];    // 34 KB: bf16 weightsT (padded)
    __shared__ int    rowptr[NPW + 1];
    __shared__ int    cur[NPW];
    __shared__ int    csr_src[EPW];        //  8 KB: src ids grouped by dst

    const int tid = threadIdx.x;
    const int w   = blockIdx.x;
    const int nodeBase = w * NPW;

    // Stage node features (coalesced) into padded f32 tile
    const float* xw = x + (size_t)nodeBase * DIM;
    for (int i = tid; i < NPW * DIM; i += 256)
        feat[(i >> 7) * FSTR + (i & 127)] = xw[i];

    // Build CSR (by dst) in LDS: count -> prefix -> fill
    if (tid < NPW) cur[tid] = 0;
    __syncthreads();
    const int* es = esrc + (size_t)w * EPW;
    const int* ed = edst + (size_t)w * EPW;
    for (int e = tid; e < EPW; e += 256)
        atomicAdd(&cur[ed[e] - nodeBase], 1);
    __syncthreads();
    if (tid == 0) {
        int s = 0;
        for (int n = 0; n < NPW; ++n) { rowptr[n] = s; s += cur[n]; }
        rowptr[NPW] = s;
    }
    __syncthreads();
    if (tid < NPW) cur[tid] = rowptr[tid];
    __syncthreads();
    for (int e = tid; e < EPW; e += 256) {
        const int d    = ed[e] - nodeBase;
        const int slot = atomicAdd(&cur[d], 1);
        csr_src[slot]  = es[e] - nodeBase;
    }
    __syncthreads();

    // ---- GIN layer 1 ----
    aggregate_csr(feat, abf, rowptr, csr_src, tid);        // abf = bf16(z1)
    gemm_relu_128(abf, w1a, wbf, b1a, abf, nullptr, tid);  // abf = relu(z1@w1a+b1a)
    gemm_relu_128(abf, w1b, wbf, b1b, abf, feat, tid);     // h1 -> abf + feat

    // ---- GIN layer 2 ----
    aggregate_csr(feat, abf, rowptr, csr_src, tid);        // abf = bf16(z2)
    gemm_relu_128(abf, w2a, wbf, b2a, abf, nullptr, tid);  // abf = relu(z2@w2a+b2a)
    gemm_relu_128(abf, w2b, wbf, b2b, abf, feat, tid);     // h2 -> feat (f32)

    // ---- mean pool over the window's 128 nodes ----
    if (tid < DIM) {
        float s = 0.f;
        for (int m = 0; m < NPW; ++m) s += feat[m * FSTR + tid];
        g[(size_t)w * DIM + tid] = s * (1.0f / (float)NPW);
    }
}

// ---------------------------------------------------------------------------
// Kernel 2: masked GRU (pack_padded semantics == iterate t < len) + head.
// One block per subject, thread j = hidden unit j. Negligible runtime cost.
// ---------------------------------------------------------------------------
__global__ __launch_bounds__(128) void gru_head_kernel(
    const float* __restrict__ g,        // [NWIN, DIM]
    const int*   __restrict__ lengths,  // [BSUBJ]
    const float* __restrict__ wih,      // [3*DIM, DIM] rows: r, z, n
    const float* __restrict__ whh,      // [3*DIM, DIM]
    const float* __restrict__ bih,      // [3*DIM]
    const float* __restrict__ bhh,      // [3*DIM]
    const float* __restrict__ wout,     // [DIM]
    const float* __restrict__ bout,     // [1]
    float* __restrict__ out)            // [BSUBJ]
{
    __shared__ float hsh[DIM], xsh[DIM], red[DIM];
    const int b = blockIdx.x;
    const int j = threadIdx.x;

    int start = 0;
    for (int i = 0; i < b; ++i) start += lengths[i];
    const int len = lengths[b];

    hsh[j] = 0.f;
    __syncthreads();

    const float bir = bih[j], biz = bih[DIM + j], bin = bih[2 * DIM + j];
    const float bhr = bhh[j], bhz = bhh[DIM + j], bhn = bhh[2 * DIM + j];
    const float* wr = wih + (size_t)j * DIM;
    const float* wz = wih + (size_t)(DIM + j) * DIM;
    const float* wn = wih + (size_t)(2 * DIM + j) * DIM;
    const float* ur = whh + (size_t)j * DIM;
    const float* uz = whh + (size_t)(DIM + j) * DIM;
    const float* un = whh + (size_t)(2 * DIM + j) * DIM;

    for (int t = 0; t < len; ++t) {
        xsh[j] = g[(size_t)(start + t) * DIM + j];
        __syncthreads();

        float rx = bir, zx = biz, nx = bin;
        float rh = bhr, zh = bhz, nh = bhn;
#pragma unroll 4
        for (int k = 0; k < DIM; ++k) {
            const float xk = xsh[k], hk = hsh[k];
            rx = fmaf(wr[k], xk, rx);
            zx = fmaf(wz[k], xk, zx);
            nx = fmaf(wn[k], xk, nx);
            rh = fmaf(ur[k], hk, rh);
            zh = fmaf(uz[k], hk, zh);
            nh = fmaf(un[k], hk, nh);
        }
        const float r  = 1.f / (1.f + __expf(-(rx + rh)));
        const float z  = 1.f / (1.f + __expf(-(zx + zh)));
        const float n  = tanhf(nx + r * nh);
        const float hn = (1.f - z) * n + z * hsh[j];
        __syncthreads();
        hsh[j] = hn;
        __syncthreads();
    }

    red[j] = hsh[j] * wout[j];
    __syncthreads();
    if (j == 0) {
        float s = bout[0];
        for (int k = 0; k < DIM; ++k) s += red[k];
        out[b] = s;
    }
}

// ---------------------------------------------------------------------------
// Launch. Input order (setup_inputs): x, edge_index, batch_idx, lengths,
// w1a, b1a, w1b, b1b, w2a, b2a, w2b, b2b, wih, whh, bih, bhh, wout, bout.
// ---------------------------------------------------------------------------
extern "C" void kernel_launch(void* const* d_in, const int* in_sizes, int n_in,
                              void* d_out, int out_size, void* d_ws, size_t ws_size,
                              hipStream_t stream) {
    const float* x       = (const float*)d_in[0];
    const int*   edge    = (const int*)d_in[1];
    /* d_in[2] = batch_idx, implied by window structure -> unused */
    const int*   lengths = (const int*)d_in[3];
    const float* w1a = (const float*)d_in[4];
    const float* b1a = (const float*)d_in[5];
    const float* w1b = (const float*)d_in[6];
    const float* b1b = (const float*)d_in[7];
    const float* w2a = (const float*)d_in[8];
    const float* b2a = (const float*)d_in[9];
    const float* w2b = (const float*)d_in[10];
    const float* b2b = (const float*)d_in[11];
    const float* wih = (const float*)d_in[12];
    const float* whh = (const float*)d_in[13];
    const float* bih = (const float*)d_in[14];
    const float* bhh = (const float*)d_in[15];
    const float* wout = (const float*)d_in[16];
    const float* bout = (const float*)d_in[17];

    float* g = (float*)d_ws;   // [NWIN * DIM] f32 window embeddings (512 KB)

    gin_window_kernel<<<NWIN, 256, 0, stream>>>(
        x, edge, edge + E_TOT,
        w1a, b1a, w1b, b1b, w2a, b2a, w2b, b2b, g);

    gru_head_kernel<<<BSUBJ, 128, 0, stream>>>(
        g, lengths, wih, whh, bih, bhh, wout, bout, (float*)d_out);
}
